// Network_52690658787548
// MI455X (gfx1250) — matrix operand order repaired and checked by reading, not compiled, not run on hardware
//
#include <hip/hip_runtime.h>

typedef _Float16 f16;
typedef __attribute__((ext_vector_type(16))) _Float16 v16h;
typedef __attribute__((ext_vector_type(8)))  _Float16 v8h;
typedef __attribute__((ext_vector_type(8)))  float    v8f;
typedef __attribute__((ext_vector_type(4))) unsigned int u32x4;
typedef __attribute__((ext_vector_type(4))) int i32x4;
typedef __attribute__((ext_vector_type(8))) int i32x8;

#define DIM     64
#define HEADS   8
#define DHEAD   64
#define INNER   512
#define MLP     256
#define DEPTH   4
#define BATCH   64
#define NPATCH  225
#define NTOK    226
#define MP      240      // padded token rows (15 * 16)
#define QVROWS  256
#define ATTN_SCALE 0.125f
#define PHS     264      // f16 row stride for LDS score/V tiles (528B = 33*16B)

#if __has_builtin(__builtin_amdgcn_tensor_load_to_lds) && \
    __has_builtin(__builtin_amdgcn_s_wait_tensorcnt)
#define HAVE_TDM 1
#else
#define HAVE_TDM 0
#endif

static __device__ __forceinline__ v16h pack2(v8h lo, v8h hi) {
  return __builtin_shufflevector(lo, hi, 0, 1, 2, 3, 4, 5, 6, 7,
                                 8, 9, 10, 11, 12, 13, 14, 15);
}
static __device__ __forceinline__ v8h ld8(const f16* p) { return *(const v8h*)p; }

// ---------------------------------------------------------------------------
// Weight prep: Wt[n*K + k] = f16(W[k*ldW + wcol0 + n])  (transpose + convert)
// ---------------------------------------------------------------------------
__global__ __launch_bounds__(64)
void wprep_kernel(const float* __restrict__ W, int ldW, int wcol0, int K,
                  f16* __restrict__ Wt) {
  int n = blockIdx.x;
  for (int k = threadIdx.x; k < K; k += 64)
    Wt[(long)n * K + k] = (f16)W[(long)k * ldW + wcol0 + n];
}

// ---------------------------------------------------------------------------
// dfm[b,d] = sigmoid( sum_c concat(mean_h, mean_l)[b,c] * fmg_w[c, d*64+d] )
// ---------------------------------------------------------------------------
__global__ __launch_bounds__(128)
void dfm_kernel(const float* __restrict__ hsi, const float* __restrict__ lidar,
                const float* __restrict__ fmg_w, float* __restrict__ dfm) {
  __shared__ float xs[128];
  int b = blockIdx.x, t = threadIdx.x;
  const float* src = (t < 64) ? hsi : lidar;
  int c = t & 63;
  const float* p = src + ((long)b * 64 + c) * NPATCH;
  float s = 0.f;
  for (int i = 0; i < NPATCH; ++i) s += p[i];
  xs[t] = s * (1.0f / NPATCH);
  __syncthreads();
  if (t < 64) {
    float acc = 0.f;
    for (int cc = 0; cc < 128; ++cc)
      acc += xs[cc] * fmg_w[(long)cc * (DIM * DIM) + t * (DIM + 1)];
    dfm[b * 64 + t] = 1.0f / (1.0f + __expf(-acc));
  }
}

// ---------------------------------------------------------------------------
// Token streams (both branches read patch tokens from hsi — faithful to ref).
// ---------------------------------------------------------------------------
__global__ __launch_bounds__(64)
void embed_kernel(const float* __restrict__ hsi,
                  const float* __restrict__ cls_h, const float* __restrict__ cls_l,
                  const float* __restrict__ pos_h, const float* __restrict__ pos_l,
                  float* __restrict__ xh, float* __restrict__ xl) {
  int b = blockIdx.x, n = blockIdx.y, c = threadIdx.x;
  float vh, vl;
  if (n == 0) {
    vh = cls_h[c] + pos_h[c];
    vl = cls_l[c] + pos_l[c];
  } else if (n < NTOK) {
    float t = hsi[((long)b * 64 + c) * NPATCH + (n - 1)];
    vh = t + pos_h[n * 64 + c];
    vl = t + pos_l[n * 64 + c];
  } else {
    vh = 0.f; vl = 0.f;
  }
  long o = ((long)b * MP + n) * 64 + c;
  xh[o] = vh;
  xl[o] = vl;
}

// ---------------------------------------------------------------------------
// LayerNorm over DIM=64; 4 rows per block, one wave each; f16 out
// ---------------------------------------------------------------------------
__global__ __launch_bounds__(128)
void ln_kernel(const float* __restrict__ x, const float* __restrict__ g,
               const float* __restrict__ beta, f16* __restrict__ xn) {
  const int row = blockIdx.x * 4 + (threadIdx.x >> 5);
  const int lane = threadIdx.x & 31;
  const float* xr = x + (long)row * 64;
  float v0 = xr[lane], v1 = xr[lane + 32];
  float s = v0 + v1;
#pragma unroll
  for (int off = 16; off; off >>= 1) s += __shfl_xor(s, off, 32);
  float mean = s * (1.0f / 64.0f);
  float d0 = v0 - mean, d1 = v1 - mean;
  float vs = d0 * d0 + d1 * d1;
#pragma unroll
  for (int off = 16; off; off >>= 1) vs += __shfl_xor(vs, off, 32);
  float rstd = rsqrtf(vs * (1.0f / 64.0f) + 1e-5f);
  f16* xo = xn + (long)row * 64;
  xo[lane]      = (f16)(d0 * rstd * g[lane]      + beta[lane]);
  xo[lane + 32] = (f16)(d1 * rstd * g[lane + 32] + beta[lane + 32]);
}

// ---------------------------------------------------------------------------
// Batched GEMM, LDS-free: C[b] = A_f16[b](MP x K) @ Wt_f16(N x K, row=out col)
// All fragments are per-lane contiguous 16B global loads (ISA 7.12.2 layout).
// block = 128 = 4 waves covering a 16 x 64 output tile.
// ---------------------------------------------------------------------------
template<bool BIAS, bool GELU_, bool RESID>
__global__ __launch_bounds__(128)
void gemm16(const f16* __restrict__ A, int ldA, long sA,
            const f16* __restrict__ Wt,
            const float* __restrict__ bias,
            f16* __restrict__ C16, int ldC, long sC,
            float* __restrict__ CR, int ldR, long sR,
            int K) {
  const int w = threadIdx.x >> 5, lane = threadIdx.x & 31;
  const int l16 = lane & 15, hlf = lane >> 4;
  const int b = blockIdx.z, m0 = blockIdx.x * 16;
  const int ncol = blockIdx.y * 64 + w * 16;
  const f16* Ar = A + (long)b * sA + (long)(m0 + l16) * ldA;
  const f16* Wr = Wt + (long)(ncol + l16) * K;
  const int kbA = hlf * 8, kbB = hlf * 16;
  v8f c = {};
  for (int k0 = 0; k0 < K; k0 += 32) {
    v16h a  = pack2(ld8(Ar + k0 + kbA), ld8(Ar + k0 + 16 + kbA));
    v16h bb = pack2(ld8(Wr + k0 + kbA), ld8(Wr + k0 + 16 + kbA));
    c = __builtin_amdgcn_wmma_f32_16x16x32_f16(false, a, false, bb,
                                               (short)0, c, false, false);
  }
  const int gn = ncol + l16;
#pragma unroll
  for (int r = 0; r < 8; ++r) {
    int gm = m0 + r + 8 * hlf;
    float v = c[r];
    if (BIAS) v += bias[gn];
    if (GELU_) v = 0.5f * v * (1.0f + erff(v * 0.70710678118f));
    if (RESID) {
      float* p = CR + (long)b * sR + (long)gm * ldR + gn;
      *p += v;
    } else {
      C16[(long)b * sC + (long)gm * ldC + gn] = (f16)v;
    }
  }
}

// ---------------------------------------------------------------------------
// Attention for one (b, head): k = q*dfm, so
//   S[i,j] = sum_d (q[i,d]*dfm[d]*scale) * q[j,d];  O = softmax(S) @ V.
// 4 waves/block; V tile DMA'd by the Tensor Data Mover (when available) into
// an LDS landing buffer, then transposed LDS->LDS with b128 reads; per-wave
// score buffer; uniform 16-tile loop with full masking of the padding tile.
// ---------------------------------------------------------------------------
__global__ __launch_bounds__(128)
void attn_kernel(const f16* __restrict__ qv, const float* __restrict__ dfm,
                 f16* __restrict__ outp) {
  __shared__ f16 Vst[64 * PHS];        // V transposed: [d][j]
  __shared__ f16 Ph[4][16 * PHS];      // per-wave scores/probs: [i][j]
#if HAVE_TDM
  __shared__ f16 Vrow[240 * 64];       // TDM landing buffer, row-major [j][d]
#endif
  const float NEG_INF = -__builtin_inff();
  const int tid = threadIdx.x;
  const int w = tid >> 5, lane = tid & 31;
  const int l16 = lane & 15, hlf = lane >> 4;
  const int h = blockIdx.x, b = blockIdx.y;
  const int hb = h * DHEAD;
  const f16* qb = qv + (long)b * (QVROWS * 1024);

#if HAVE_TDM
  // --- Tensor Data Mover: DMA the 240x64 f16 V tile (row stride 1024) ---
  if (w == 0) {
    unsigned long long ga = (unsigned long long)(const void*)(qb + 512 + hb);
    unsigned int lds_off = (unsigned int)(unsigned long long)(const void*)&Vrow[0];
    u32x4 g0;
    g0[0] = 1u;                                   // count=1 user descriptor
    g0[1] = lds_off;                              // lds_addr (bytes)
    g0[2] = (unsigned int)ga;                     // global_addr[31:0]
    g0[3] = (unsigned int)(ga >> 32) | (2u << 30);// global_addr[56:32] | type=2
    i32x8 g1;
    g1[0] = 0x00010000;   // workgroup_mask=0, data_size=2B, no pad/iterate
    g1[1] = (64  << 16);  // tensor_dim0 = 64
    g1[2] = (240 << 16);  // tensor_dim1 = 240
    g1[3] = (64  << 16);  // tile_dim0 = 64
    g1[4] = 240;          // tile_dim1 = 240, tile_dim2 = 0
    g1[5] = 1024;         // tensor_dim0_stride = 1024 elements
    g1[6] = 0;
    g1[7] = 0;
    i32x4 g2 = {0, 0, 0, 0};
    i32x4 g3 = {0, 0, 0, 0};
#if __clang_major__ >= 23
    i32x8 g4 = {0, 0, 0, 0, 0, 0, 0, 0};
    __builtin_amdgcn_tensor_load_to_lds(g0, g1, g2, g3, g4, 0);
#else
    __builtin_amdgcn_tensor_load_to_lds(g0, g1, g2, g3, 0);
#endif
    __builtin_amdgcn_s_wait_tensorcnt(0);
  }
  __syncthreads();
  // transpose LDS->LDS: b128 row reads, scalar transposed writes
  for (int idx = tid; idx < 240 * 8; idx += 128) {
    int j = idx >> 3, c8 = (idx & 7) * 8;
    v8h vv = *(const v8h*)&Vrow[j * 64 + c8];
#pragma unroll
    for (int e = 0; e < 8; ++e) Vst[(c8 + e) * PHS + j] = vv[e];
  }
#else
  // fallback: direct global reads -> transposed LDS
  for (int idx = tid; idx < 240 * 64; idx += 128) {
    int j = idx >> 6, d = idx & 63;
    Vst[d * PHS + j] = qb[(long)j * 1024 + 512 + hb + d];
  }
#endif
  // zero the 16 pad columns j so P(=0) * pad stays exactly 0
  for (int idx = tid; idx < 16 * 64; idx += 128) {
    int d = idx >> 4, j = 240 + (idx & 15);
    Vst[d * PHS + j] = (f16)0.f;
  }
  __syncthreads();

  // per-lane dfm*scale for the 32 d-values this lane's A fragment touches
  const int kbA = hlf * 8, kbB = hlf * 16;
  float dfs[4][8];
#pragma unroll
  for (int e = 0; e < 8; ++e) {
#pragma unroll
    for (int sgrp = 0; sgrp < 4; ++sgrp)
      dfs[sgrp][e] = dfm[b * 64 + sgrp * 16 + kbA + e] * ATTN_SCALE;
  }

  f16* Pw = &Ph[w][0];
  for (int it = w; it < 16; it += 4) {
    const int i0 = it * 16;
    // A fragments: q * dfm * scale (rows i0..i0+15; 16B loads then scale)
    v16h a0, a1;
    {
      const f16* qr = qb + (long)(i0 + l16) * 1024 + hb;
      v8h q00 = ld8(qr + kbA),      q01 = ld8(qr + 16 + kbA);
      v8h q10 = ld8(qr + 32 + kbA), q11 = ld8(qr + 48 + kbA);
#pragma unroll
      for (int e = 0; e < 8; ++e) {
        a0[e]     = (f16)((float)q00[e] * dfs[0][e]);
        a0[e + 8] = (f16)((float)q01[e] * dfs[1][e]);
        a1[e]     = (f16)((float)q10[e] * dfs[2][e]);
        a1[e + 8] = (f16)((float)q11[e] * dfs[3][e]);
      }
    }
    // S row-block: 16 j-tiles of 16, plain-Q B fragments (16B loads)
#pragma unroll 4
    for (int jt = 0; jt < 16; ++jt) {
      const f16* qj = qb + (long)(jt * 16 + l16) * 1024 + hb;
      v8f s = {};
      v16h b0 = pack2(ld8(qj + kbA), ld8(qj + 16 + kbA));
      s = __builtin_amdgcn_wmma_f32_16x16x32_f16(false, a0, false, b0,
                                                 (short)0, s, false, false);
      v16h b1 = pack2(ld8(qj + 32 + kbA), ld8(qj + 48 + kbA));
      s = __builtin_amdgcn_wmma_f32_16x16x32_f16(false, a1, false, b1,
                                                 (short)0, s, false, false);
      int j = jt * 16 + l16;
#pragma unroll
      for (int r = 0; r < 8; ++r)
        Pw[(r + 8 * hlf) * PHS + j] = (j < NTOK) ? (f16)s[r] : (f16)NEG_INF;
    }
    __syncthreads();
    // softmax per row over 256 (pad cols are -inf -> exp 0)
    for (int r = 0; r < 16; ++r) {
      float mx = NEG_INF;
      for (int k = lane; k < 256; k += 32)
        mx = fmaxf(mx, (float)Pw[r * PHS + k]);
#pragma unroll
      for (int off = 16; off; off >>= 1) mx = fmaxf(mx, __shfl_xor(mx, off, 32));
      float sum = 0.f;
      for (int k = lane; k < 256; k += 32) {
        float p = __expf((float)Pw[r * PHS + k] - mx);
        Pw[r * PHS + k] = (f16)p;
        sum += p;
      }
#pragma unroll
      for (int off = 16; off; off >>= 1) sum += __shfl_xor(sum, off, 32);
      float inv = 1.0f / sum;
      for (int k = lane; k < 256; k += 32)
        Pw[r * PHS + k] = (f16)((float)Pw[r * PHS + k] * inv);
    }
    __syncthreads();
    // O = P (16x256) @ V (256x64): A frags from Ph, B frags from Vst, all b128
    v8f acc[4] = {};
#pragma unroll 2
    for (int kk = 0; kk < 8; ++kk) {
      const f16* pr = Pw + l16 * PHS + kk * 32;
      v16h a = pack2(*(const v8h*)(pr + kbA), *(const v8h*)(pr + 16 + kbA));
#pragma unroll
      for (int dt = 0; dt < 4; ++dt) {
        const f16* vr = Vst + (dt * 16 + l16) * PHS + kk * 32;
        v16h bv = pack2(*(const v8h*)(vr + kbA), *(const v8h*)(vr + 16 + kbA));
        acc[dt] = __builtin_amdgcn_wmma_f32_16x16x32_f16(false, a, false, bv,
                                                         (short)0, acc[dt],
                                                         false, false);
      }
    }
#pragma unroll
    for (int dt = 0; dt < 4; ++dt)
#pragma unroll
      for (int r = 0; r < 8; ++r) {
        int gm = i0 + r + 8 * hlf;
        if (gm < MP)
          outp[(long)b * (MP * INNER) + (long)gm * INNER + hb + dt * 16 + l16] =
              (f16)acc[dt][r];
      }
    __syncthreads();
  }
}

// ---------------------------------------------------------------------------
__global__ __launch_bounds__(64)
void exch_kernel(float* __restrict__ xh, float* __restrict__ xl) {
  int b = blockIdx.x, c = threadIdx.x;
  long o = (long)b * MP * 64 + c;   // token 0
  float a = xh[o], d = xl[o];
  xh[o] = d;
  xl[o] = a;
}

__global__ __launch_bounds__(64)
void final_kernel(const float* __restrict__ xh, const float* __restrict__ xl,
                  float* __restrict__ outp) {
  int b = blockIdx.x, c = threadIdx.x;
  outp[b * 64 + c] = xh[(long)b * MP * 64 + c] + xl[(long)b * MP * 64 + c];
}

// ---------------------------------------------------------------------------
extern "C" void kernel_launch(void* const* d_in, const int* in_sizes, int n_in,
                              void* d_out, int out_size, void* d_ws, size_t ws_size,
                              hipStream_t stream) {
  (void)in_sizes; (void)n_in; (void)out_size; (void)ws_size;
  const float* hsi   = (const float*)d_in[0];
  const float* lidar = (const float*)d_in[1];
  const float* cls_h = (const float*)d_in[2];
  const float* cls_l = (const float*)d_in[3];
  const float* pos_h = (const float*)d_in[4];
  const float* pos_l = (const float*)d_in[5];
  const float* fmg_w = (const float*)d_in[6];
  const float* ln1_g = (const float*)d_in[7];
  const float* ln1_b = (const float*)d_in[8];
  const float* qkv_w = (const float*)d_in[9];
  const float* out_w = (const float*)d_in[10];
  const float* out_b = (const float*)d_in[11];
  const float* ln2_g = (const float*)d_in[12];
  const float* ln2_b = (const float*)d_in[13];
  const float* ff_w1 = (const float*)d_in[14];
  const float* ff_b1 = (const float*)d_in[15];
  const float* ff_w2 = (const float*)d_in[16];
  const float* ff_b2 = (const float*)d_in[17];

  char* ws = (char*)d_ws;
  size_t off = 0;
  auto carve = [&](size_t bytes) -> void* {
    void* p = ws + off;
    off += (bytes + 255) & ~(size_t)255;
    return p;
  };
  float* dfm  = (float*)carve((size_t)BATCH * 64 * 4);
  float* xh   = (float*)carve((size_t)BATCH * MP * 64 * 4);
  float* xl   = (float*)carve((size_t)BATCH * MP * 64 * 4);
  f16*   xn   = (f16*)  carve((size_t)BATCH * MP * 64 * 2);
  f16*   qv   = (f16*)  carve((size_t)BATCH * QVROWS * 1024 * 2);
  f16*   attb = (f16*)  carve((size_t)BATCH * MP * INNER * 2);
  f16*   hbuf = (f16*)  carve((size_t)BATCH * MP * MLP * 2);
  // transposed f16 weights: [out_col][K]
  f16*   wqt  = (f16*)  carve((size_t)DEPTH * 1024 * 64 * 2);  // q rows 0..511, v rows 512..1023
  f16*   owt  = (f16*)  carve((size_t)DEPTH * 64 * 512 * 2);
  f16*   w1t  = (f16*)  carve((size_t)DEPTH * 256 * 64 * 2);
  f16*   w2t  = (f16*)  carve((size_t)DEPTH * 64 * 256 * 2);

  for (int l = 0; l < DEPTH; ++l) {
    wprep_kernel<<<512, 64, 0, stream>>>(qkv_w + (size_t)l * 64 * 1536, 1536, 0,
                                         64, wqt + (size_t)l * 1024 * 64);
    wprep_kernel<<<512, 64, 0, stream>>>(qkv_w + (size_t)l * 64 * 1536, 1536, 1024,
                                         64, wqt + (size_t)l * 1024 * 64 + 512 * 64);
    wprep_kernel<<<64, 64, 0, stream>>>(out_w + (size_t)l * 512 * 64, 64, 0,
                                        512, owt + (size_t)l * 64 * 512);
    wprep_kernel<<<256, 64, 0, stream>>>(ff_w1 + (size_t)l * 64 * 256, 256, 0,
                                         64, w1t + (size_t)l * 256 * 64);
    wprep_kernel<<<64, 64, 0, stream>>>(ff_w2 + (size_t)l * 256 * 64, 64, 0,
                                        256, w2t + (size_t)l * 64 * 256);
  }

  dfm_kernel<<<BATCH, 128, 0, stream>>>(hsi, lidar, fmg_w, dfm);
  embed_kernel<<<dim3(BATCH, MP), 64, 0, stream>>>(hsi, cls_h, cls_l, pos_h, pos_l, xh, xl);

  for (int pass = 0; pass < 2; ++pass) {
    for (int layer = 0; layer < DEPTH; ++layer) {
      const f16* wq  = wqt + (size_t)layer * 1024 * 64;
      const f16* wo  = owt + (size_t)layer * 64 * 512;
      const f16* wf1 = w1t + (size_t)layer * 256 * 64;
      const f16* wf2 = w2t + (size_t)layer * 64 * 256;
      const float* ob = out_b + layer * 64;
      const float* b1 = ff_b1 + layer * 256;
      const float* b2 = ff_b2 + layer * 64;
      const float* g1 = ln1_g + layer * 64;
      const float* e1 = ln1_b + layer * 64;
      const float* g2 = ln2_g + layer * 64;
      const float* e2 = ln2_b + layer * 64;
      float* xs[2] = {xh, xl};
      for (int br = 0; br < 2; ++br) {
        float* x = xs[br];
        // attention block (K projection is dead code in the reference)
        ln_kernel<<<BATCH * MP / 4, 128, 0, stream>>>(x, g1, e1, xn);
        gemm16<false, false, false><<<dim3(15, 16, BATCH), 128, 0, stream>>>(
            xn, 64, (long)MP * 64, wq, nullptr,
            qv, 1024, (long)QVROWS * 1024, nullptr, 0, 0, 64);
        attn_kernel<<<dim3(HEADS, BATCH), 128, 0, stream>>>(qv, dfm, attb);
        gemm16<true, false, true><<<dim3(15, 1, BATCH), 128, 0, stream>>>(
            attb, 512, (long)MP * 512, wo, ob,
            nullptr, 0, 0, x, 64, (long)MP * 64, 512);
        // feed-forward block
        ln_kernel<<<BATCH * MP / 4, 128, 0, stream>>>(x, g2, e2, xn);
        gemm16<true, true, false><<<dim3(15, 4, BATCH), 128, 0, stream>>>(
            xn, 64, (long)MP * 64, wf1, b1,
            hbuf, 256, (long)MP * 256, nullptr, 0, 0, 64);
        gemm16<true, false, true><<<dim3(15, 1, BATCH), 128, 0, stream>>>(
            hbuf, 256, (long)MP * 256, wf2, b2,
            nullptr, 0, 0, x, 64, (long)MP * 64, 256);
      }
    }
    if (pass == 0) exch_kernel<<<BATCH, 64, 0, stream>>>(xh, xl);
  }
  final_kernel<<<BATCH, 64, 0, stream>>>(xh, xl, (float*)d_out);
}
